// CasMVSNet_15006615733590
// MI455X (gfx1250) — matrix operand order repaired and checked by reading, not compiled, not run on hardware
//
#include <hip/hip_runtime.h>
#include <hip/hip_bf16.h>

typedef __attribute__((ext_vector_type(16))) _Float16 v16h;
typedef __attribute__((ext_vector_type(8)))  _Float16 v8h;
typedef __attribute__((ext_vector_type(8)))  float    v8f;

typedef unsigned int v4u  __attribute__((ext_vector_type(4)));
typedef int          v8i_ __attribute__((ext_vector_type(8)));
typedef int          v4i_ __attribute__((ext_vector_type(4)));

#define BN_EPS 1e-5f

#if __has_builtin(__builtin_amdgcn_tensor_load_to_lds) && \
    __has_builtin(__builtin_amdgcn_s_wait_tensorcnt)
#define HAVE_TDM 1
#else
#define HAVE_TDM 0
#endif

// ---------------------------------------------------------------------------
// TDM: DMA a 2-D f32 tile [tileRows][rowLen] (row stride rowLen) from global
// memory into LDS at byte offset ldsOff. OOB rows (>= rowsRemaining) read 0.
// Descriptor packing per CDNA5 ISA 8.3/8.4 (D# group0 128b, group1 256b).
// This toolchain's builtin takes 6 args (g0, g1, g2, g3, g_extra, cpol).
// ---------------------------------------------------------------------------
__device__ __forceinline__ void tdm_load_2d_f32(unsigned ldsOff, const float* gptr,
                                                int rowLen, int rowsRemaining,
                                                int tileRows) {
#if HAVE_TDM
  unsigned long long ga = (unsigned long long)(uintptr_t)gptr;
  v4u g0;
  g0.x = 1u;                                        // count=1, user descriptor
  g0.y = ldsOff;                                    // lds_addr (bytes)
  g0.z = (unsigned)(ga & 0xFFFFFFFFu);              // global_addr[31:0]
  g0.w = (unsigned)((ga >> 32) & 0x1FFFFFFu)        // global_addr[56:32]
       | (2u << 30);                                // type=2 ("image")
  v8i_ g1;
  g1[0] = (int)(2u << 16);                          // data_size=2 -> 4 bytes
  g1[1] = (int)((rowLen & 0xFFFF) << 16);           // tensor_dim0[15:0]
  g1[2] = (int)(((unsigned)(rowLen >> 16) & 0xFFFFu) |
                (((unsigned)rowsRemaining & 0xFFFFu) << 16)); // tdim0 hi, tdim1 lo
  g1[3] = (int)(((unsigned)(rowsRemaining >> 16) & 0xFFFFu) |
                (((unsigned)rowLen & 0xFFFFu) << 16));        // tdim1 hi, tile_dim0
  g1[4] = (int)(tileRows & 0xFFFF);                 // tile_dim1 (tile_dim2=0)
  g1[5] = rowLen;                                   // tensor_dim0_stride[31:0]
  g1[6] = 0;
  g1[7] = 0;
  v4i_ z4 = {0, 0, 0, 0};
  v8i_ z8 = {0, 0, 0, 0, 0, 0, 0, 0};
  __builtin_amdgcn_tensor_load_to_lds(g0, g1, z4, z4, z8, 0);
  __builtin_amdgcn_s_wait_tensorcnt(0);
#else
  (void)ldsOff; (void)gptr; (void)rowLen; (void)rowsRemaining; (void)tileRows;
#endif
}

// Assemble a 16-half fragment from two contiguous 16-byte LDS runs.
__device__ __forceinline__ v16h frag_from_lds_split(const _Float16* p) {
  v8h lo = *(const v8h*)p;
  v8h hi = *(const v8h*)(p + 16);
  return __builtin_shufflevector(lo, hi, 0, 1, 2, 3, 4, 5, 6, 7,
                                 8, 9, 10, 11, 12, 13, 14, 15);
}

// ---------------------------------------------------------------------------
// Bilinear resize, half-pixel centers, edge clamp. Handles any channel count.
// ---------------------------------------------------------------------------
__global__ void k_resize(const float* __restrict__ in, float* __restrict__ out,
                         int C, int Hi, int Wi, int Ho, int Wo) {
  int idx = blockIdx.x * blockDim.x + threadIdx.x;
  int total = C * Ho * Wo;
  if (idx >= total) return;
  int xo = idx % Wo;
  int yo = (idx / Wo) % Ho;
  int c  = idx / (Wo * Ho);
  float sx = (xo + 0.5f) * (float)Wi / (float)Wo - 0.5f;
  float sy = (yo + 0.5f) * (float)Hi / (float)Ho - 0.5f;
  float x0f = floorf(sx), y0f = floorf(sy);
  float wx = sx - x0f, wy = sy - y0f;
  int x0 = (int)x0f, y0 = (int)y0f;
  int x1 = x0 + 1, y1 = y0 + 1;
  x0 = min(max(x0, 0), Wi - 1); x1 = min(max(x1, 0), Wi - 1);
  y0 = min(max(y0, 0), Hi - 1); y1 = min(max(y1, 0), Hi - 1);
  const float* p = in + (size_t)c * Hi * Wi;
  float v00 = p[y0 * Wi + x0], v01 = p[y0 * Wi + x1];
  float v10 = p[y1 * Wi + x0], v11 = p[y1 * Wi + x1];
  out[idx] = (1.f - wy) * ((1.f - wx) * v00 + wx * v01)
           +         wy * ((1.f - wx) * v10 + wx * v11);
}

// ---------------------------------------------------------------------------
// feature_net layer 1: conv2d 3x3 (Cin=3 -> C) + bias + BN + ReLU, scalar.
// ---------------------------------------------------------------------------
__global__ void k_conv1_bn_relu(const float* __restrict__ in, const float* __restrict__ w,
                                const float* __restrict__ b,
                                const float* __restrict__ g, const float* __restrict__ bb,
                                const float* __restrict__ mn, const float* __restrict__ vr,
                                float* __restrict__ out, int C, int H, int W) {
  int idx = blockIdx.x * blockDim.x + threadIdx.x;
  if (idx >= H * W) return;
  int x = idx % W, y = idx / W;
  for (int co = 0; co < C; ++co) {
    float s = b[co];
    for (int ci = 0; ci < 3; ++ci)
      for (int ky = 0; ky < 3; ++ky) {
        int yy = y + ky - 1;
        if (yy < 0 || yy >= H) continue;
        for (int kx = 0; kx < 3; ++kx) {
          int xx = x + kx - 1;
          if (xx < 0 || xx >= W) continue;
          s += in[(ci * H + yy) * W + xx] * w[((co * 3 + ci) * 3 + ky) * 3 + kx];
        }
      }
    float sc = g[co] * rsqrtf(vr[co] + BN_EPS);
    float o  = s * sc + (bb[co] - mn[co] * sc);
    out[(co * H + y) * W + x] = fmaxf(o, 0.f);
  }
}

// ---------------------------------------------------------------------------
// WMMA implicit-GEMM conv2d 3x3, Cin -> Cout (+bias). One wave32 per block.
// Tile: M = 16 consecutive x at fixed y, N = 16 couts, K = Cin*9 (k = r*Cin+ci,
// Cin is a power of two so decompose = shift/mask).
// LDS: patch[16][Kp] f16 (im2row A), wtile[16][Kp] f16 (B), wraw[16][K] f32
// filled by the Tensor Data Mover. Fragments are contiguous LDS reads:
//   A(lane m): halves at m*Kp + k0 + 8*laneHi (+0..7) and +16 (+0..7)
//   B(lane n): halves at n*Kp + k0 + 8*laneHi (+0..7) and +16 (+0..7)
// ---------------------------------------------------------------------------
__global__ void k_conv2d_wmma(const float* __restrict__ in, const float* __restrict__ w,
                              const float* __restrict__ bias, float* __restrict__ out,
                              int Cin, int Cout, int H, int W, int cs, int Kp) {
  extern __shared__ char smem[];
  _Float16* patch = (_Float16*)smem;             // [16][Kp]
  _Float16* wtile = patch + 16 * Kp;             // [16][Kp]
  float*    wraw  = (float*)(wtile + 16 * Kp);   // [16][K]

  const int lane   = threadIdx.x;
  const int m      = lane & 15;                  // A row / B col / D col
  const int laneHi = lane >> 4;
  const int tilesX = W >> 4;
  const int x0 = (blockIdx.x % tilesX) << 4;
  const int y  = blockIdx.x / tilesX;
  const int n0 = blockIdx.y << 4;
  const int K  = Cin * 9;
  const int nB = n0 + m;

#if HAVE_TDM
  tdm_load_2d_f32((unsigned)(uintptr_t)wraw, w + (size_t)n0 * K, K, Cout - n0, 16);
#endif

  // A: im2row patch into LDS (coalesced over m for fixed k).
  for (int k = laneHi; k < Kp; k += 2) {
    float va = 0.f;
    if (k < K) {
      int ci = k & (Cin - 1);
      int r  = k >> cs;
      int ry = r / 3, rx = r - ry * 3;
      int yy = y + ry - 1, xx = x0 + m + rx - 1;
      if (yy >= 0 && yy < H && xx >= 0 && xx < W)
        va = in[(ci * H + yy) * W + xx];
    }
    patch[m * Kp + k] = (_Float16)va;
  }
  // B: reorder + f32->f16 weights into LDS.
  for (int k = laneHi; k < Kp; k += 2) {
    float vb = 0.f;
    if (k < K && nB < Cout) {
      int ci = k & (Cin - 1);
      int r  = k >> cs;
#if HAVE_TDM
      vb = wraw[m * K + ci * 9 + r];
#else
      vb = w[((size_t)nB * Cin + ci) * 9 + r];
#endif
    }
    wtile[m * Kp + k] = (_Float16)vb;
  }

  v8f acc = {};
  for (int k0 = 0; k0 < Kp; k0 += 32) {
    v16h a = frag_from_lds_split(patch + m * Kp + k0 + (laneHi << 3));
    v16h b = frag_from_lds_split(wtile + m * Kp + k0 + (laneHi << 3));
    acc = __builtin_amdgcn_wmma_f32_16x16x32_f16(false, a, false, b,
                                                 (short)0, acc, false, false);
  }
  if (nB < Cout) {
    float bv = bias[nB];
#pragma unroll
    for (int r = 0; r < 8; ++r) {
      int mm = r + (laneHi << 3);
      out[(nB * H + y) * W + x0 + mm] = acc[r] + bv;
    }
  }
}

// ---------------------------------------------------------------------------
// WMMA implicit-GEMM conv3d 3x3x3, Cin -> Cout (+bias, optional ReLU).
// Same structure as conv2d; K = Cin*27, tile pixels along x at fixed (d,y).
// ---------------------------------------------------------------------------
__global__ void k_conv3d_wmma(const float* __restrict__ in, const float* __restrict__ w,
                              const float* __restrict__ bias, float* __restrict__ out,
                              int Cin, int Cout, int D, int H, int W,
                              int relu, int cs, int Kp) {
  extern __shared__ char smem[];
  _Float16* patch = (_Float16*)smem;             // [16][Kp]
  _Float16* wtile = patch + 16 * Kp;             // [16][Kp]
  float*    wraw  = (float*)(wtile + 16 * Kp);   // [16][K]

  const int lane   = threadIdx.x;
  const int m      = lane & 15;
  const int laneHi = lane >> 4;
  const int tilesX = W >> 4;
  const int x0 = (blockIdx.x % tilesX) << 4;
  int t = blockIdx.x / tilesX;
  const int y = t % H;
  const int d = t / H;
  const int n0 = blockIdx.y << 4;
  const int K  = Cin * 27;
  const int nB = n0 + m;

#if HAVE_TDM
  tdm_load_2d_f32((unsigned)(uintptr_t)wraw, w + (size_t)n0 * K, K, Cout - n0, 16);
#endif

  for (int k = laneHi; k < Kp; k += 2) {
    float va = 0.f;
    if (k < K) {
      int ci = k & (Cin - 1);
      int r  = k >> cs;                          // r < 27
      int rz = r / 9, r2 = r - rz * 9;
      int ry = r2 / 3, rx = r2 - ry * 3;
      int zz = d + rz - 1, yy = y + ry - 1, xx = x0 + m + rx - 1;
      if (zz >= 0 && zz < D && yy >= 0 && yy < H && xx >= 0 && xx < W)
        va = in[((size_t)(ci * D + zz) * H + yy) * W + xx];
    }
    patch[m * Kp + k] = (_Float16)va;
  }
  for (int k = laneHi; k < Kp; k += 2) {
    float vb = 0.f;
    if (k < K && nB < Cout) {
      int ci = k & (Cin - 1);
      int r  = k >> cs;
#if HAVE_TDM
      vb = wraw[m * K + ci * 27 + r];
#else
      vb = w[((size_t)nB * Cin + ci) * 27 + r];
#endif
    }
    wtile[m * Kp + k] = (_Float16)vb;
  }

  v8f acc = {};
  for (int k0 = 0; k0 < Kp; k0 += 32) {
    v16h a = frag_from_lds_split(patch + m * Kp + k0 + (laneHi << 3));
    v16h b = frag_from_lds_split(wtile + m * Kp + k0 + (laneHi << 3));
    acc = __builtin_amdgcn_wmma_f32_16x16x32_f16(false, a, false, b,
                                                 (short)0, acc, false, false);
  }
  if (nB < Cout) {
    float bv = bias[nB];
#pragma unroll
    for (int r = 0; r < 8; ++r) {
      int mm = r + (laneHi << 3);
      float o = acc[r] + bv;
      if (relu) o = fmaxf(o, 0.f);
      out[((size_t)(nB * D + d) * H + y) * W + x0 + mm] = o;
    }
  }
}

// ---------------------------------------------------------------------------
// Plane-sweep cost volume: per (d,y,x) homography-warp both source feature
// maps, bilinear sample with zero-outside-border weights, accumulate mean
// squared difference over views per channel.
// ---------------------------------------------------------------------------
__global__ void k_cost(const float* __restrict__ refF, const float* __restrict__ srcF,
                       const float* __restrict__ proj, const float* __restrict__ pd,
                       float dmin, float dstep, float* __restrict__ cost,
                       int C, int D, int H, int W) {
  int idx = blockIdx.x * blockDim.x + threadIdx.x;
  int HW = H * W;
  if (idx >= D * HW) return;
  int x = idx % W, y = (idx / W) % H, d = idx / HW;
  float dep = dmin + d * dstep + (pd ? pd[y * W + x] : 0.f);

  float tw[2][4];
  int   ti[2][4];
#pragma unroll
  for (int v = 0; v < 2; ++v) {
    const float* Pm = proj + v * 12;
    float fx = (float)x, fy = (float)y;
    float b0 = Pm[0] * fx + Pm[1] * fy + Pm[2];
    float b1 = Pm[4] * fx + Pm[5] * fy + Pm[6];
    float b2 = Pm[8] * fx + Pm[9] * fy + Pm[10];
    float pc0 = dep * b0 + Pm[3];
    float pc1 = dep * b1 + Pm[7];
    float pc2 = dep * b2 + Pm[11];
    float z  = pc2 + 1e-8f;
    float gx = pc0 / z, gy = pc1 / z;
    float x0f = floorf(gx), y0f = floorf(gy);
    float wx = gx - x0f, wy = gy - y0f;
    int xi = (int)x0f, yi = (int)y0f;
#pragma unroll
    for (int j = 0; j < 4; ++j) {
      int xj = xi + (j & 1), yj = yi + (j >> 1);
      float wj = ((j & 1) ? wx : 1.f - wx) * ((j >> 1) ? wy : 1.f - wy);
      bool valid = (xj >= 0) && (xj <= W - 1) && (yj >= 0) && (yj <= H - 1);
      int xc = min(max(xj, 0), W - 1), yc = min(max(yj, 0), H - 1);
      tw[v][j] = valid ? wj : 0.f;
      ti[v][j] = yc * W + xc;
    }
  }

  size_t CHW = (size_t)C * HW;
  for (int c = 0; c < C; ++c) {
    float r = refF[(size_t)c * HW + y * W + x];
    float a = 0.f;
#pragma unroll
    for (int v = 0; v < 2; ++v) {
      const float* SF = srcF + (size_t)v * CHW + (size_t)c * HW;
      float s = SF[ti[v][0]] * tw[v][0] + SF[ti[v][1]] * tw[v][1]
              + SF[ti[v][2]] * tw[v][2] + SF[ti[v][3]] * tw[v][3];
      float df = r - s;
      a += df * df;
    }
    cost[((size_t)c * D + d) * HW + y * W + x] = a * 0.5f;  // /V, V=2
  }
}

// ---------------------------------------------------------------------------
// Final 3D conv, Cout = 1, scalar (K <= 8*27, a WMMA tile would be ~94% pad).
// ---------------------------------------------------------------------------
__global__ void k_conv3d_c1(const float* __restrict__ in, const float* __restrict__ w,
                            const float* __restrict__ b, float* __restrict__ out,
                            int Cin, int D, int H, int W) {
  int idx = blockIdx.x * blockDim.x + threadIdx.x;
  int HW = H * W;
  if (idx >= D * HW) return;
  int x = idx % W, y = (idx / W) % H, d = idx / HW;
  float s = b[0];
  for (int ci = 0; ci < Cin; ++ci)
    for (int kz = 0; kz < 3; ++kz) {
      int zz = d + kz - 1;
      if (zz < 0 || zz >= D) continue;
      for (int ky = 0; ky < 3; ++ky) {
        int yy = y + ky - 1;
        if (yy < 0 || yy >= H) continue;
        for (int kx = 0; kx < 3; ++kx) {
          int xx = x + kx - 1;
          if (xx < 0 || xx >= W) continue;
          s += in[((size_t)(ci * D + zz) * H + yy) * W + xx]
             * w[ci * 27 + kz * 9 + ky * 3 + kx];
        }
      }
    }
  out[idx] = s;
}

// ---------------------------------------------------------------------------
// Softmax over depth + expectation + max-prob confidence.
// ---------------------------------------------------------------------------
__global__ void k_softmax_depth(const float* __restrict__ pv, const float* __restrict__ pd,
                                float dmin, float dstep,
                                float* __restrict__ dout, float* __restrict__ conf,
                                int D, int HW) {
  int idx = blockIdx.x * blockDim.x + threadIdx.x;
  if (idx >= HW) return;
  float base = (pd ? pd[idx] : 0.f) + dmin;
  float mx = -3.4e38f;
  for (int d = 0; d < D; ++d) mx = fmaxf(mx, pv[(size_t)d * HW + idx]);
  float se = 0.f, acc = 0.f, pm = 0.f;
  for (int d = 0; d < D; ++d) {
    float e = __expf(pv[(size_t)d * HW + idx] - mx);
    se  += e;
    acc += e * (base + d * dstep);
    pm   = fmaxf(pm, e);
  }
  dout[idx] = acc / se;
  conf[idx] = pm / se;
}

// ---------------------------------------------------------------------------
// Host orchestration
// ---------------------------------------------------------------------------
static inline int ilog2i(int v) { int s = 0; while ((1 << s) < v) ++s; return s; }
static inline int round32(int v) { return (v + 31) & ~31; }

extern "C" void kernel_launch(void* const* d_in, const int* in_sizes, int n_in,
                              void* d_out, int out_size, void* d_ws, size_t ws_size,
                              hipStream_t stream) {
  (void)in_sizes; (void)n_in; (void)out_size; (void)ws_size;
  const float* refImg256 = (const float*)d_in[0];   // (1,3,256,256)
  const float* srcImg256 = (const float*)d_in[1];   // (2,1,3,256,256)
  const float* proj      = (const float*)d_in[2];   // (2,3,4)
  enum { FW1, FB1, BN_G, BN_B, BN_M, BN_V, FW2, FB2, RW1, RB1, RW2, RB2, RW3, RB3 };
  auto P = [&](int s, int k) -> const float* {
    return (const float*)d_in[3 + s * 14 + k];
  };

  // Workspace layout (floats), sized for the per-stage maxima:
  float* ws    = (float*)d_ws;
  float* refS0 = ws;                     //  3*64*64
  float* srcS0 = refS0 + 3 * 4096;       //  6*64*64
  float* fh    = srcS0 + 6 * 4096;       //  max C*H*W      = 32*65536
  float* refF  = fh    + 2097152;        //  max C*H*W
  float* srcF  = refF  + 2097152;        //  2 * max C*H*W
  float* pdBuf = srcF  + 2 * 2097152;    //  256*256
  float* costB = pdBuf + 65536;          //  max C*D*H*W     = 16*32*65536
  float* h1    = costB + 33554432;       //  max (C/2)*D*H*W =  8*32*65536
  float* h2    = h1    + 16777216;       //  max (C/4)*D*H*W =  4*32*65536
  float* pvB   = h2    + 8388608;        //  max D*H*W       = 32*65536

  float* outF = (float*)d_out;
  const int dOff[3] = {0, 8192, 139264};
  const int cOff[3] = {4096, 73728, 204800};

  // Stage-0 input pyramids (256 -> 64).
  {
    int n = 3 * 64 * 64;
    k_resize<<<(n + 255) / 256, 256, 0, stream>>>(refImg256, refS0, 3, 256, 256, 64, 64);
    n = 6 * 64 * 64;
    k_resize<<<(n + 255) / 256, 256, 0, stream>>>(srcImg256, srcS0, 6, 256, 256, 64, 64);
  }

  for (int s = 0; s < 3; ++s) {
    const int C  = 8 << s;
    const int Dn = (s == 0) ? 48 : ((s == 1) ? 32 : 8);
    const int H  = (s == 0) ? 64 : 256, W = H;
    const int HW = H * W;
    const float* rImg = (s == 0) ? refS0 : refImg256;
    const float* sImg = (s == 0) ? srcS0 : srcImg256;

    // feature_net on ref + 2 source images.
    {
      const int K2 = C * 9, Kp2 = round32(K2);
      const size_t sh2 = (size_t)16 * Kp2 * 2 * 2 + (size_t)16 * K2 * 4;
      for (int img = 0; img < 3; ++img) {
        const float* src = (img == 0) ? rImg : (sImg + (size_t)(img - 1) * 3 * HW);
        float*       dst = (img == 0) ? refF : (srcF + (size_t)(img - 1) * C * HW);
        k_conv1_bn_relu<<<(HW + 255) / 256, 256, 0, stream>>>(
            src, P(s, FW1), P(s, FB1), P(s, BN_G), P(s, BN_B), P(s, BN_M), P(s, BN_V),
            fh, C, H, W);
        dim3 g2((W >> 4) * H, (C + 15) >> 4);
        k_conv2d_wmma<<<g2, 32, sh2, stream>>>(fh, P(s, FW2), P(s, FB2), dst,
                                               C, C, H, W, ilog2i(C), Kp2);
      }
    }

    // Depth hypotheses: dep(d) = base + dmin + d*dstep.
    const float* pd = nullptr;
    float dmin, dstep;
    if (s == 0) {
      dmin  = 425.0f;
      dstep = 48.0f * 2.5f / 47.0f;          // linspace(425, 425+48*2.5, 48)
    } else {
      int Hp = (s == 1) ? 64 : 256;          // upsample previous depth
      int n = 256 * 256;
      k_resize<<<(n + 255) / 256, 256, 0, stream>>>(outF + dOff[s - 1], pdBuf,
                                                    1, Hp, Hp, 256, 256);
      pd    = pdBuf;
      dmin  = -(float)Dn * 2.5f / 4.0f;                 // offs[0]
      dstep = (float)Dn * 2.5f / (2.0f * (Dn - 1));     // offs step
    }

    // Cost volume.
    {
      int n = Dn * HW;
      k_cost<<<(n + 255) / 256, 256, 0, stream>>>(refF, srcF, proj, pd, dmin, dstep,
                                                  costB, C, Dn, H, W);
    }

    // 3D regularization: conv3d(ReLU) x2 via WMMA, then Cout=1 scalar conv.
    {
      const int Ka = C * 27, Kpa = round32(Ka);
      const size_t sha = (size_t)16 * Kpa * 2 * 2 + (size_t)16 * Ka * 4;
      dim3 g1((W >> 4) * H * Dn, ((C / 2) + 15) >> 4);
      k_conv3d_wmma<<<g1, 32, sha, stream>>>(costB, P(s, RW1), P(s, RB1), h1,
                                             C, C / 2, Dn, H, W, 1, ilog2i(C), Kpa);
      const int Kb = (C / 2) * 27, Kpb = round32(Kb);
      const size_t shb = (size_t)16 * Kpb * 2 * 2 + (size_t)16 * Kb * 4;
      dim3 g2((W >> 4) * H * Dn, ((C / 4) + 15) >> 4);
      k_conv3d_wmma<<<g2, 32, shb, stream>>>(h1, P(s, RW2), P(s, RB2), h2,
                                             C / 2, C / 4, Dn, H, W, 1,
                                             ilog2i(C / 2), Kpb);
      int n = Dn * HW;
      k_conv3d_c1<<<(n + 255) / 256, 256, 0, stream>>>(h2, P(s, RW3), P(s, RB3),
                                                       pvB, C / 4, Dn, H, W);
    }

    // Softmax depth regression + confidence straight into d_out.
    k_softmax_depth<<<(HW + 255) / 256, 256, 0, stream>>>(
        pvB, pd, dmin, dstep, outF + dOff[s], outF + cOff[s], Dn, HW);
  }
}
